// ChebyshevKAN_14078902796758
// MI455X (gfx1250) — compile-verified, matrix operand order checked by reading, not run on hardware
//
#include <hip/hip_runtime.h>
#include <hip/hip_bf16.h>
#include <cstdint>

// ---------------------------------------------------------------------------
// ChebyshevKAN forward for MI455X (gfx1250, wave32, WMMA bf16)
//   layer: y = sum_d T_d(tanh(a)) @ coeff[:,:,d]^T + a @ base_w^T + bias
// Implemented as a single K'-extended GEMM (7 "planes" per 32-wide K chunk)
// using v_wmma_f32_16x16x32_bf16 with f32 accumulation.
// ---------------------------------------------------------------------------

typedef __attribute__((ext_vector_type(16))) __bf16 v16bf;
typedef __attribute__((ext_vector_type(8)))  __bf16 v8bf;
typedef __attribute__((ext_vector_type(8)))  float  v8f;
typedef __attribute__((ext_vector_type(4)))  float  v4f;

constexpr int BSZ    = 16384;
constexpr int DIM0   = 1024;
constexpr int DIM1   = 1024;
constexpr int DIM2   = 512;
constexpr int NPLANE = 7;     // T0..T5 + raw-activation (base_w) plane
constexpr int KC     = 32;    // WMMA bf16 K
constexpr int BM     = 128;   // block tile M
constexpr int BN     = 128;   // block tile N

// ---------------------------------------------------------------------------
// Pack f32 weights into bf16 WMMA B-fragment layout.
// Fragment = one (K=32) x (N=16) tile: 32 lanes x 16 bf16 (1 KB).
// Element (k, n) of a fragment lives at:
//   lane    = (n % 16) + 16 * ((k % 16) >= 8)
//   halfidx = (k / 16) * 8 + (k % 8)
// Fragment order: ((kc * 7 + plane) * (O/16) + ntile)
// ---------------------------------------------------------------------------
__global__ __launch_bounds__(256)
void pack_weights(const float* __restrict__ coeff,    // [O][I][6]
                  const float* __restrict__ base_w,   // [O][I]
                  __bf16* __restrict__ dst,           // 7*I*O bf16, fragment-major
                  int I, int O)
{
    long long idx   = (long long)blockIdx.x * blockDim.x + threadIdx.x;
    long long total = (long long)NPLANE * I * O;
    if (idx >= total) return;

    int       within  = (int)(idx & 511);      // element inside 512-elem fragment
    long long frag    = idx >> 9;
    int       lane    = within >> 4;
    int       halfidx = within & 15;
    int       ntiles  = O >> 4;
    int       nt      = (int)(frag % ntiles);
    long long rest    = frag / ntiles;
    int       p       = (int)(rest % NPLANE);
    int       kc      = (int)(rest / NPLANE);

    int hi  = lane >> 4;
    int n16 = lane & 15;
    int grp = halfidx >> 3;
    int kk8 = halfidx & 7;
    int k   = grp * 16 + hi * 8 + kk8;
    int i   = kc * KC + k;
    int o   = nt * 16 + n16;

    float v = (p < 6) ? coeff[((long long)o * I + i) * 6 + p]
                      : base_w[(long long)o * I + i];
    dst[idx] = (__bf16)v;
}

// ---------------------------------------------------------------------------
// Fused Chebyshev GEMM:
//   out[b,o] = sum_i sum_p plane_p(act[b,i]) * Wpack[p,i,o] + bias[o]
// plane_0..5 = Chebyshev T_d(tanh(act)), plane_6 = act.
// Block: 256 threads (8 wave32), 128x128 output tile; wave grid 2x4,
// each wave: 4 (M) x 2 (N) accumulators of 16x16 f32 (v8f each).
// A planes staged in LDS in WMMA A-fragment order (ISA 7.12.2 16-bit layout):
//   element (m, k): lane = m%16 + 16*((k%16)>=8), halfidx = (k/16)*8 + k%8
// ---------------------------------------------------------------------------
__global__ __launch_bounds__(256)
void cheb_gemm(const float*  __restrict__ act,    // [B][I] f32
               const __bf16* __restrict__ wpack,  // packed 7*I*O bf16
               const float*  __restrict__ bias,   // [O]
               float*        __restrict__ out,    // [B][O] f32
               int I, int O, int pre_tanh)
{
    __shared__ __bf16 smem[NPLANE * 8 * 512];     // 7 planes x 8 mtiles x 1KB = 56 KB

    const int t    = threadIdx.x;
    const int lane = t & 31;
    const int wave = t >> 5;
    const int wm   = wave >> 2;                   // 0..1  (64 rows each)
    const int wn   = wave & 3;                    // 0..3  (32 cols each)
    const int bm   = blockIdx.x * BM;
    const int bn   = blockIdx.y * BN;
    const int ntileO = O >> 4;

    v8f acc[4][2];
    const v8f vzero = {0.f, 0.f, 0.f, 0.f, 0.f, 0.f, 0.f, 0.f};
#pragma unroll
    for (int mt = 0; mt < 4; ++mt)
#pragma unroll
        for (int nt = 0; nt < 2; ++nt) acc[mt][nt] = vzero;

    // staging assignment: thread -> (row r in tile, k-half h)
    const int r  = t >> 1;                        // 0..127
    const int h  = t & 1;                         // 0..1 (k 0..15 / 16..31)
    const int ml = r & 15;
    const int mt8 = r >> 4;
    const float* arow = act + (long long)(bm + r) * I + h * 16;

    const int nchunk = I / KC;
    for (int kc = 0; kc < nchunk; ++kc) {
        // ---- stage A planes into LDS (pre-swizzled fragment order) ----
        float xv[16], xn[16];
        {
            const float* src = arow + (long long)kc * KC;
            v4f a0 = *(const v4f*)(src + 0);
            v4f a1 = *(const v4f*)(src + 4);
            v4f a2 = *(const v4f*)(src + 8);
            v4f a3 = *(const v4f*)(src + 12);
#pragma unroll
            for (int j = 0; j < 4; ++j) {
                xv[j]      = a0[j];
                xv[4 + j]  = a1[j];
                xv[8 + j]  = a2[j];
                xv[12 + j] = a3[j];
            }
        }
        if (pre_tanh) {
#pragma unroll
            for (int j = 0; j < 16; ++j) xv[j] = tanhf(xv[j]);   // input normalization
        }
#pragma unroll
        for (int j = 0; j < 16; ++j) xn[j] = tanhf(xv[j]);       // cheb normalization

        auto store_plane = [&](int p, const float* vals) {
            v8bf lo, hi;
#pragma unroll
            for (int q = 0; q < 8; ++q) lo[q] = (__bf16)vals[q];
#pragma unroll
            for (int q = 0; q < 8; ++q) hi[q] = (__bf16)vals[8 + q];
            __bf16* basep = &smem[(p * 8 + mt8) * 512];
            *(v8bf*)(basep + ml * 16 + h * 8)        = lo;   // lanes 0..15  (k%16 < 8)
            *(v8bf*)(basep + (ml + 16) * 16 + h * 8) = hi;   // lanes 16..31 (k%16 >= 8)
        };

        float tp[16], tc[16];
#pragma unroll
        for (int j = 0; j < 16; ++j) { tp[j] = 1.0f; tc[j] = xn[j]; }
        store_plane(0, tp);           // T0 = 1
        store_plane(1, tc);           // T1 = xn
#pragma unroll
        for (int d = 2; d <= 5; ++d) {
            float tn[16];
#pragma unroll
            for (int j = 0; j < 16; ++j) tn[j] = 2.0f * xn[j] * tc[j] - tp[j];
            store_plane(d, tn);
#pragma unroll
            for (int j = 0; j < 16; ++j) { tp[j] = tc[j]; tc[j] = tn[j]; }
        }
        store_plane(6, xv);           // base_w plane uses raw activation
        __syncthreads();

        // ---- WMMA over 7 planes ----
        const __bf16* wbase = wpack + (long long)kc * NPLANE * ntileO * 512;
        if (kc + 1 < nchunk)          // pull next chunk's weights toward L0/L2
            __builtin_prefetch((const void*)(wbase + (long long)NPLANE * ntileO * 512
                                             + lane * 16), 0, 1);
#pragma unroll
        for (int p = 0; p < NPLANE; ++p) {
            v16bf afrag[4];
#pragma unroll
            for (int mt = 0; mt < 4; ++mt)
                afrag[mt] = *(const v16bf*)&smem[(p * 8 + wm * 4 + mt) * 512 + lane * 16];
            v16bf bfrag[2];
#pragma unroll
            for (int nt = 0; nt < 2; ++nt) {
                long long fid = (long long)p * ntileO + ((bn >> 4) + wn * 2 + nt);
                bfrag[nt] = *(const v16bf*)(wbase + fid * 512 + lane * 16);
            }
#pragma unroll
            for (int mt = 0; mt < 4; ++mt)
#pragma unroll
                for (int nt = 0; nt < 2; ++nt)
                    acc[mt][nt] = __builtin_amdgcn_wmma_f32_16x16x32_bf16(
                        false, afrag[mt], false, bfrag[nt],
                        (short)0, acc[mt][nt], false, false);
        }
        __syncthreads();
    }

    // ---- epilogue: +bias, store f32 (C/D layout: VGPR rr -> M=rr / M=rr+8) ----
    const int mrow0 = bm + wm * 64;
    const int ncol0 = bn + wn * 32;
    const int nlo   = lane & 15;
    const int mhalf = (lane >> 4) * 8;
#pragma unroll
    for (int mt = 0; mt < 4; ++mt)
#pragma unroll
        for (int nt = 0; nt < 2; ++nt) {
            const int   col  = ncol0 + nt * 16 + nlo;
            const float bcol = bias[col];
#pragma unroll
            for (int rr = 0; rr < 8; ++rr) {
                const int row = mrow0 + mt * 16 + mhalf + rr;
                out[(long long)row * O + col] = acc[mt][nt][rr] + bcol;
            }
        }
}

// ---------------------------------------------------------------------------
// In-place LayerNorm + SiLU over rows of h [B][D], D = 1024, 256 threads/row.
// ---------------------------------------------------------------------------
__global__ __launch_bounds__(256)
void ln_silu(float* __restrict__ hbuf,
             const float* __restrict__ gamma,
             const float* __restrict__ beta, int D)
{
    __shared__ float red[16];
    const int t = threadIdx.x;
    float* rowp = hbuf + (long long)blockIdx.x * D;

    v4f v = *(const v4f*)(rowp + t * 4);
    float s = 0.f, s2 = 0.f;
#pragma unroll
    for (int j = 0; j < 4; ++j) { s += v[j]; s2 += v[j] * v[j]; }

    // wave32 reduction
#pragma unroll
    for (int off = 16; off > 0; off >>= 1) {
        s  += __shfl_down(s,  off, 32);
        s2 += __shfl_down(s2, off, 32);
    }
    if ((t & 31) == 0) { red[(t >> 5) * 2] = s; red[(t >> 5) * 2 + 1] = s2; }
    __syncthreads();

    float S = 0.f, S2 = 0.f;
#pragma unroll
    for (int w = 0; w < 8; ++w) { S += red[w * 2]; S2 += red[w * 2 + 1]; }

    const float mu   = S / (float)D;
    const float var  = S2 / (float)D - mu * mu;
    const float rinv = rsqrtf(var + 1e-5f);
#pragma unroll
    for (int j = 0; j < 4; ++j) {
        const int c = t * 4 + j;
        float y = (v[j] - mu) * rinv * gamma[c] + beta[c];
        v[j] = y / (1.0f + __expf(-y));          // SiLU
    }
    *(v4f*)(rowp + t * 4) = v;
}

// ---------------------------------------------------------------------------
// Launcher
// inputs: x, coeff1, base_w1, bias1, ln_gamma, ln_beta, coeff2, base_w2, bias2
// ws layout: h[B*D1] f32 (64 MB) | wp1 bf16 7*D0*D1 (14 MB) | wp2 bf16 7*D1*D2 (7 MB)
// ---------------------------------------------------------------------------
extern "C" void kernel_launch(void* const* d_in, const int* in_sizes, int n_in,
                              void* d_out, int out_size, void* d_ws, size_t ws_size,
                              hipStream_t stream)
{
    (void)in_sizes; (void)n_in; (void)out_size; (void)ws_size;

    const float* x       = (const float*)d_in[0];
    const float* coeff1  = (const float*)d_in[1];
    const float* base_w1 = (const float*)d_in[2];
    const float* bias1   = (const float*)d_in[3];
    const float* gamma   = (const float*)d_in[4];
    const float* beta    = (const float*)d_in[5];
    const float* coeff2  = (const float*)d_in[6];
    const float* base_w2 = (const float*)d_in[7];
    const float* bias2   = (const float*)d_in[8];
    float* outp = (float*)d_out;

    float*  hbuf = (float*)d_ws;
    __bf16* wp1  = (__bf16*)((char*)d_ws + (size_t)BSZ * DIM1 * sizeof(float));
    __bf16* wp2  = wp1 + (size_t)NPLANE * DIM0 * DIM1;

    // pack weights to bf16 fragment layout (stays resident in 192 MB L2)
    long long tot1 = (long long)NPLANE * DIM0 * DIM1;
    pack_weights<<<(unsigned)((tot1 + 255) / 256), 256, 0, stream>>>(coeff1, base_w1, wp1, DIM0, DIM1);
    long long tot2 = (long long)NPLANE * DIM1 * DIM2;
    pack_weights<<<(unsigned)((tot2 + 255) / 256), 256, 0, stream>>>(coeff2, base_w2, wp2, DIM1, DIM2);

    // layer 1: h = cheb(tanh(x)) + bias1
    cheb_gemm<<<dim3(BSZ / BM, DIM1 / BN), 256, 0, stream>>>(x, wp1, bias1, hbuf, DIM0, DIM1, 1);
    // LayerNorm + SiLU (in place)
    ln_silu<<<BSZ, 256, 0, stream>>>(hbuf, gamma, beta, DIM1);
    // layer 2: out = cheb(h) + bias2
    cheb_gemm<<<dim3(BSZ / BM, DIM2 / BN), 256, 0, stream>>>(hbuf, wp2, bias2, outp, DIM1, DIM2, 0);
}
